// GTLayer_5411658793076
// MI455X (gfx1250) — compile-verified, matrix operand order checked by reading, not run on hardware
//
#include <hip/hip_runtime.h>
#include <hip/hip_bf16.h>

#define N_NODES 20000
#define N_EDGES 640000
#define HIDDEN 256
#define HEADS 8
#define HEAD_DIM 32
#define VOCAB 119
#define N_FEATS 9

typedef __attribute__((ext_vector_type(16))) __bf16 v16bf;
typedef __attribute__((ext_vector_type(8)))  float  v8f;

__device__ __forceinline__ __bf16 f2bf(float f) {
    union { float f; unsigned u; } a; a.f = f;
    unsigned r = a.u + 0x7FFFu + ((a.u >> 16) & 1u);  // round-nearest-even
    unsigned short hv = (unsigned short)(r >> 16);
    __bf16 o; __builtin_memcpy(&o, &hv, 2); return o;
}

// A-matrix 16x32 bf16 per-lane pattern (ISA 7.12.2). B^T = W loaded with the
// same pattern (lane = N/output-channel row of W, regs = K).
__device__ __forceinline__ v16bf load_tile(const __bf16* base, int ld, int row, int half, int kbase) {
    const __bf16* p = base + (size_t)row * ld + kbase;
    v16bf r;
#pragma unroll
    for (int j = 0; j < 16; j += 2) {
        int k = (j < 8) ? (half * 8 + j) : (16 + half * 8 + (j - 8));
        r[j]     = p[k];
        r[j + 1] = p[k + 1];
    }
    return r;
}

// ---------------- kernel 1: weights f32 -> bf16 -------------------------
__global__ __launch_bounds__(256) void cvt_w_kernel(
    const float* __restrict__ qw, const float* __restrict__ kw,
    const float* __restrict__ vw, const float* __restrict__ ow,
    __bf16* __restrict__ w3, __bf16* __restrict__ owb)
{
    int t = blockIdx.x * 256 + threadIdx.x;
    if (t < HIDDEN * HIDDEN) {
        w3[t]                       = f2bf(qw[t]);
        w3[HIDDEN * HIDDEN + t]     = f2bf(kw[t]);
        w3[2 * HIDDEN * HIDDEN + t] = f2bf(vw[t]);
        owb[t]                      = f2bf(ow[t]);
    }
}

// ---------------- kernel 2: atom embedding sum --------------------------
__global__ __launch_bounds__(256) void embed_kernel(
    const int* __restrict__ X, const float* __restrict__ tab, __bf16* __restrict__ h)
{
    __shared__ int xs[N_FEATS];
    int n = blockIdx.x;
    if (threadIdx.x < N_FEATS) xs[threadIdx.x] = X[n * N_FEATS + threadIdx.x];
    __syncthreads();
    int c = threadIdx.x;
    float acc = 0.f;
#pragma unroll
    for (int f = 0; f < N_FEATS; f++)
        acc += tab[((size_t)f * VOCAB + xs[f]) * HIDDEN + c];
    h[(size_t)n * HIDDEN + c] = f2bf(acc);
}

// ---------------- kernel 3: fused Q/K/V projection (WMMA) ---------------
__global__ __launch_bounds__(256) void qkv_gemm_kernel(
    const __bf16* __restrict__ h, const __bf16* __restrict__ w3,
    const float* __restrict__ qb, const float* __restrict__ kb, const float* __restrict__ vb,
    float* __restrict__ q, float* __restrict__ k, float* __restrict__ v)
{
    const int NT_N = HIDDEN / 16;               // 16 tiles along output dim
    int tile = blockIdx.x * 8 + (threadIdx.x >> 5);
    int tm = tile / NT_N, tn = tile % NT_N;
    if (tm >= N_NODES / 16) return;
    int lane = threadIdx.x & 31;
    int mr = lane & 15, half = lane >> 4;
    const __bf16* A  = h  + (size_t)tm * 16 * HIDDEN;
    const __bf16* Bq = w3 + (size_t)tn * 16 * HIDDEN;
    const __bf16* Bk = Bq + (size_t)HIDDEN * HIDDEN;
    const __bf16* Bv = Bk + (size_t)HIDDEN * HIDDEN;
    v8f aq = {}, ak = {}, av = {};
#pragma unroll
    for (int kk = 0; kk < HIDDEN; kk += 32) {
        v16bf at = load_tile(A, HIDDEN, mr, half, kk);
        v16bf bt = load_tile(Bq, HIDDEN, mr, half, kk);
        aq = __builtin_amdgcn_wmma_f32_16x16x32_bf16(false, at, false, bt, (short)0, aq, false, false);
        bt = load_tile(Bk, HIDDEN, mr, half, kk);
        ak = __builtin_amdgcn_wmma_f32_16x16x32_bf16(false, at, false, bt, (short)0, ak, false, false);
        bt = load_tile(Bv, HIDDEN, mr, half, kk);
        av = __builtin_amdgcn_wmma_f32_16x16x32_bf16(false, at, false, bt, (short)0, av, false, false);
    }
    int col = tn * 16 + mr;                     // D layout: N = lane & 15
    int mhi = half * 8;                         // D layout: M = r + 8*(lane>>4)
    float qbv = qb[col], kbv = kb[col], vbv = vb[col];
    const float scale = 0.17677669529663687f;   // HEAD_DIM^-0.5
#pragma unroll
    for (int r = 0; r < 8; r++) {
        size_t off = (size_t)(tm * 16 + mhi + r) * HIDDEN + col;
        q[off] = (aq[r] + qbv) * scale;
        k[off] = ak[r] + kbv;
        v[off] = av[r] + vbv;
    }
}

// ---------------- kernel 4: CSR row_ptr via binary search ---------------
__global__ __launch_bounds__(256) void rowptr_kernel(const int* __restrict__ row, int* __restrict__ rp)
{
    int n = blockIdx.x * 256 + threadIdx.x;
    if (n > N_NODES) return;
    int lo = 0, hi = N_EDGES;
    while (lo < hi) { int mid = (lo + hi) >> 1; if (row[mid] < n) lo = mid + 1; else hi = mid; }
    rp[n] = lo;
}

// ---------------- kernel 5: SDDMM (per edge, per head dot) --------------
__global__ __launch_bounds__(256) void sddmm_kernel(
    const int* __restrict__ row, const int* __restrict__ col,
    const float* __restrict__ q, const float* __restrict__ k, float* __restrict__ s)
{
    int t = blockIdx.x * 256 + threadIdx.x;
    if (t >= N_EDGES * HEADS) return;
    int e = t >> 3, hd = t & 7;
    const float4* qp = (const float4*)(q + (size_t)row[e] * HIDDEN + hd * HEAD_DIM);
    const float4* kp = (const float4*)(k + (size_t)col[e] * HIDDEN + hd * HEAD_DIM);
    float acc = 0.f;
#pragma unroll
    for (int i = 0; i < HEAD_DIM / 4; i++) {
        float4 a = qp[i], b = kp[i];
        acc += a.x * b.x + a.y * b.y + a.z * b.z + a.w * b.w;
    }
    s[t] = acc;
}

// ---------------- kernel 6: per-(node,head) segment max + exp-sum -------
__global__ __launch_bounds__(256) void segmz_kernel(
    const float* __restrict__ s, const int* __restrict__ rp,
    float* __restrict__ m, float* __restrict__ z)
{
    int t = blockIdx.x * 256 + threadIdx.x;
    if (t >= N_NODES * HEADS) return;
    int n = t >> 3, hd = t & 7;
    int e0 = rp[n], e1 = rp[n + 1];
    float mx = -3.402823466e38f;
    for (int e = e0; e < e1; e++) mx = fmaxf(mx, s[e * HEADS + hd]);
    float zz = 0.f;
    for (int e = e0; e < e1; e++) zz += __expf(s[e * HEADS + hd] - mx);
    m[t] = mx; z[t] = zz;
}

// ---------------- kernel 7: normalize scores -> attn (in place) ---------
__global__ __launch_bounds__(256) void attn_kernel(
    const int* __restrict__ row, const float* __restrict__ m, const float* __restrict__ z,
    float* __restrict__ s)
{
    int t = blockIdx.x * 256 + threadIdx.x;
    if (t >= N_EDGES * HEADS) return;
    int e = t >> 3, hd = t & 7;
    int r = row[e];
    s[t] = __expf(s[t] - m[r * HEADS + hd]) / z[r * HEADS + hd];
}

// ---------------- kernel 8: SPMM aggregation (wave == head) -------------
__global__ __launch_bounds__(256) void spmm_kernel(
    const int* __restrict__ col, const int* __restrict__ rp,
    const float* __restrict__ attn, const float* __restrict__ v, __bf16* __restrict__ agg)
{
    int n = blockIdx.x;
    int c = threadIdx.x;
    int hd = c >> 5;                 // HEAD_DIM == wave32 width
    int e0 = rp[n], e1 = rp[n + 1];
    float acc = 0.f;
    for (int e = e0; e < e1; e++)
        acc += attn[e * HEADS + hd] * v[(size_t)col[e] * HIDDEN + c];
    agg[(size_t)n * HIDDEN + c] = f2bf(acc);
}

// ---------------- kernel 9: output projection (WMMA) --------------------
__global__ __launch_bounds__(256) void out_gemm_kernel(
    const __bf16* __restrict__ agg, const __bf16* __restrict__ ow,
    const float* __restrict__ ob, float* __restrict__ out)
{
    const int NT_N = HIDDEN / 16;
    int tile = blockIdx.x * 8 + (threadIdx.x >> 5);
    int tm = tile / NT_N, tn = tile % NT_N;
    if (tm >= N_NODES / 16) return;
    int lane = threadIdx.x & 31;
    int mr = lane & 15, half = lane >> 4;
    const __bf16* A = agg + (size_t)tm * 16 * HIDDEN;
    const __bf16* B = ow  + (size_t)tn * 16 * HIDDEN;
    v8f acc = {};
#pragma unroll
    for (int kk = 0; kk < HIDDEN; kk += 32) {
        v16bf at = load_tile(A, HIDDEN, mr, half, kk);
        v16bf bt = load_tile(B, HIDDEN, mr, half, kk);
        acc = __builtin_amdgcn_wmma_f32_16x16x32_bf16(false, at, false, bt, (short)0, acc, false, false);
    }
    int col = tn * 16 + mr;
    int mhi = half * 8;
    float obv = ob[col];
#pragma unroll
    for (int r = 0; r < 8; r++)
        out[(size_t)(tm * 16 + mhi + r) * HIDDEN + col] = acc[r] + obv;
}

static inline size_t align_up(size_t x) { return (x + 255) & ~(size_t)255; }

extern "C" void kernel_launch(void* const* d_in, const int* in_sizes, int n_in,
                              void* d_out, int out_size, void* d_ws, size_t ws_size,
                              hipStream_t stream) {
    const int*   X    = (const int*)d_in[0];
    const int*   erow = (const int*)d_in[1];
    const int*   ecol = (const int*)d_in[2];
    const float* emb  = (const float*)d_in[3];
    const float* qw   = (const float*)d_in[4];
    const float* qb   = (const float*)d_in[5];
    const float* kw   = (const float*)d_in[6];
    const float* kb   = (const float*)d_in[7];
    const float* vw   = (const float*)d_in[8];
    const float* vb   = (const float*)d_in[9];
    const float* ow   = (const float*)d_in[10];
    const float* ob   = (const float*)d_in[11];
    float* out = (float*)d_out;

    // workspace carve-up (~104 MB)
    char* ws = (char*)d_ws;
    size_t cur = 0;
    __bf16* h_bf   = (__bf16*)(ws + cur); cur = align_up(cur + (size_t)N_NODES * HIDDEN * 2);
    __bf16* w3_bf  = (__bf16*)(ws + cur); cur = align_up(cur + (size_t)3 * HIDDEN * HIDDEN * 2);
    __bf16* ow_bf  = (__bf16*)(ws + cur); cur = align_up(cur + (size_t)HIDDEN * HIDDEN * 2);
    float*  q      = (float*)(ws + cur);  cur = align_up(cur + (size_t)N_NODES * HIDDEN * 4);
    float*  k      = (float*)(ws + cur);  cur = align_up(cur + (size_t)N_NODES * HIDDEN * 4);
    float*  v      = (float*)(ws + cur);  cur = align_up(cur + (size_t)N_NODES * HIDDEN * 4);
    float*  s      = (float*)(ws + cur);  cur = align_up(cur + (size_t)N_EDGES * HEADS * 4);
    float*  m      = (float*)(ws + cur);  cur = align_up(cur + (size_t)N_NODES * HEADS * 4);
    float*  z      = (float*)(ws + cur);  cur = align_up(cur + (size_t)N_NODES * HEADS * 4);
    int*    rp     = (int*)(ws + cur);    cur = align_up(cur + (size_t)(N_NODES + 1) * 4);
    __bf16* agg_bf = (__bf16*)(ws + cur); cur = align_up(cur + (size_t)N_NODES * HIDDEN * 2);
    (void)ws_size; (void)in_sizes; (void)n_in; (void)out_size;

    const int gemm_blocks = (N_NODES / 16) * (HIDDEN / 16) / 8;   // 2500

    cvt_w_kernel<<<(HIDDEN * HIDDEN + 255) / 256, 256, 0, stream>>>(qw, kw, vw, ow, w3_bf, ow_bf);
    embed_kernel<<<N_NODES, 256, 0, stream>>>(X, emb, h_bf);
    qkv_gemm_kernel<<<gemm_blocks, 256, 0, stream>>>(h_bf, w3_bf, qb, kb, vb, q, k, v);
    rowptr_kernel<<<(N_NODES + 1 + 255) / 256, 256, 0, stream>>>(erow, rp);
    sddmm_kernel<<<(N_EDGES * HEADS + 255) / 256, 256, 0, stream>>>(erow, ecol, q, k, s);
    segmz_kernel<<<(N_NODES * HEADS + 255) / 256, 256, 0, stream>>>(s, rp, m, z);
    attn_kernel<<<(N_EDGES * HEADS + 255) / 256, 256, 0, stream>>>(erow, m, z, s);
    spmm_kernel<<<N_NODES, 256, 0, stream>>>(ecol, rp, s, v, agg_bf);
    out_gemm_kernel<<<gemm_blocks, 256, 0, stream>>>(agg_bf, ow_bf, ob, out);
}
